// GPSLayer_31997506356064
// MI455X (gfx1250) — matrix operand order repaired and checked by reading, not compiled, not run on hardware
//
#include <hip/hip_runtime.h>

// ---------------- problem constants (from reference) ----------------
#define NN 20000      // nodes
#define NE 320000     // edges
#define DD 128        // dim
#define LL 500        // nodes per graph
#define LP 512        // padded L for attention K/V
#define BB 40         // graphs
#define HH 8          // heads
#define BH (BB*HH)    // 320

// ---------------- WMMA types / helpers ----------------
typedef __attribute__((ext_vector_type(16))) __bf16 v16bf;
typedef __attribute__((ext_vector_type(8)))  __bf16 v8bf;
typedef __attribute__((ext_vector_type(8)))  float  v8f;

__device__ __forceinline__ __bf16 f2bf(float f) {
  union { float f; unsigned u; } a; a.f = f;
  unsigned r = (a.u + 0x7fffu + ((a.u >> 16) & 1u)) >> 16;   // RNE
  union { unsigned short s; __bf16 b; } o; o.s = (unsigned short)r;
  return o.b;
}

// A fragment: 16x32 bf16 from row-major [16][lda] (contiguous K layout:
// lanes 0-15 row M, K=0..15; lanes 16-31 row M, K=16..31)
__device__ __forceinline__ v16bf load_A(const __bf16* base, int lda, int k0, int lane) {
  const __bf16* p = base + (size_t)(lane & 15) * lda + k0 + ((lane >> 4) << 4);
  v16bf a;
#pragma unroll
  for (int i = 0; i < 16; ++i) a[i] = p[i];
  return a;
}

// B fragment from pre-packed weights: one contiguous 32B load per lane.
// P layout: [((nc*KC + kc)*32 + lane)*16 + i]
__device__ __forceinline__ v16bf load_Bp(const __bf16* P, int KC, int kc, int nc, int lane) {
  return *(const v16bf*)(P + (((size_t)nc * KC + kc) * 32 + lane) * 16);
}

// C/D layout: lanes 0-15 N=lane, VGPR j -> M=j; lanes 16-31 N=lane-16, M=j+8
__device__ __forceinline__ void store_C(float* base, int ldc, int n0, v8f c, int lane) {
  float* p = base + (size_t)((lane >> 4) << 3) * ldc + n0 + (lane & 15);
#pragma unroll
  for (int j = 0; j < 8; ++j) p[(size_t)j * ldc] = c[j];
}

#define WMMA(a, b, c) \
  __builtin_amdgcn_wmma_f32_16x16x32_bf16(false, (a), false, (b), (short)0, (c), false, false)

// LayerNorm over 16 rows of 128 in LDS (in place)
__device__ __forceinline__ void ln_rows(float (*T)[DD], const float* g, const float* b, int tid) {
  __syncthreads();
  if (tid < 16) {
    float m = 0.f;
    for (int c = 0; c < DD; ++c) m += T[tid][c];
    m *= (1.f / DD);
    float v = 0.f;
    for (int c = 0; c < DD; ++c) { float d = T[tid][c] - m; v += d * d; }
    v *= (1.f / DD);
    float rs = rsqrtf(v + 1e-5f);
    for (int c = 0; c < DD; ++c) T[tid][c] = (T[tid][c] - m) * rs * g[c] + b[c];
  }
  __syncthreads();
}

// ---------------- prep kernels ----------------
__global__ void k_cvt_bf16(const float* __restrict__ s, __bf16* __restrict__ d, int n) {
  int i = blockIdx.x * 256 + threadIdx.x;
  if (i < n) d[i] = f2bf(s[i]);
}
__global__ void k_zero_f32(float* p, int n) {
  int i = blockIdx.x * 256 + threadIdx.x;
  if (i < n) p[i] = 0.f;
}
__global__ void k_zero_bf16(__bf16* p, int n) {
  int i = blockIdx.x * 256 + threadIdx.x;
  if (i < n) p[i] = f2bf(0.f);
}
// pack weight W [K][Nw] (row-major f32) into WMMA B-fragment order (bf16)
__global__ void k_pack_b(const float* __restrict__ W, __bf16* __restrict__ P, int K, int Nw) {
  const int idx = blockIdx.x * 256 + threadIdx.x;
  if (idx >= K * Nw) return;
  const int i    = idx & 15;
  const int lane = (idx >> 4) & 31;
  const int blk  = idx >> 9;
  const int KC   = K >> 5;
  const int kc   = blk % KC, nc = blk / KC;
  const int k = kc * 32 + ((lane >> 4) << 4) + i;
  const int n = nc * 16 + (lane & 15);
  P[idx] = f2bf(W[(size_t)k * Nw + n]);
}

// ---------------- 1) edge MLP + scatter-add ----------------
// msg = relu([x[dst]|x[src]|edge_attr] @ e_w1 + b1) @ e_w2 + b2 ; agg[dst] += msg
__global__ __launch_bounds__(256) void k_edge(
    const __bf16* __restrict__ xbf, const float* __restrict__ eattr,
    const int* __restrict__ eidx,
    const __bf16* __restrict__ w1, const float* __restrict__ b1,
    const __bf16* __restrict__ w2, const float* __restrict__ b2,
    float* __restrict__ agg) {
  __shared__ __bf16 A1[16][3 * DD];
  __shared__ __bf16 Hs[16][DD];
  const int tile = blockIdx.x, tid = threadIdx.x;
  const int lane = tid & 31, wave = tid >> 5;
  const int* srcv = eidx;        // edge_index[0] = src (x_j)
  const int* dstv = eidx + NE;   // edge_index[1] = dst (x_i)
  // stage A tile: 16 edges x 384 (x[dst] | x[src] | edge_attr), 16B chunks/thread
  {
    const int r  = tid >> 4;            // row 0..15
    const int cb = (tid & 15) * 8;      // 8 contiguous bf16 = 16 bytes
    const int e  = tile * 16 + r;
    const __bf16* xd = xbf + (size_t)dstv[e] * DD;
    const __bf16* xs = xbf + (size_t)srcv[e] * DD;
    const float*  ea = eattr + (size_t)e * DD;
    if (e + 16 < NE) __builtin_prefetch(eattr + (size_t)(e + 16) * DD, 0, 0);
    *(v8bf*)&A1[r][cb]      = *(const v8bf*)(xd + cb);
    *(v8bf*)&A1[r][DD + cb] = *(const v8bf*)(xs + cb);
    v8bf t;
#pragma unroll
    for (int i = 0; i < 8; ++i) t[i] = f2bf(ea[cb + i]);
    *(v8bf*)&A1[r][2 * DD + cb] = t;
  }
  __syncthreads();
  v8f acc = {};
#pragma unroll
  for (int kk = 0; kk < 12; ++kk)
    acc = WMMA(load_A(&A1[0][0], 3 * DD, kk * 32, lane), load_Bp(w1, 12, kk, wave, lane), acc);
  {
    const int n = wave * 16 + (lane & 15), mb = (lane >> 4) << 3;
    const float bias = b1[n];
#pragma unroll
    for (int j = 0; j < 8; ++j) {
      float v = acc[j] + bias;
      Hs[mb + j][n] = f2bf(v > 0.f ? v : 0.f);
    }
  }
  __syncthreads();
  v8f acc2 = {};
#pragma unroll
  for (int kk = 0; kk < 4; ++kk)
    acc2 = WMMA(load_A(&Hs[0][0], DD, kk * 32, lane), load_Bp(w2, 4, kk, wave, lane), acc2);
  {
    const int n = wave * 16 + (lane & 15), mb = (lane >> 4) << 3;
    const float bias = b2[n];
#pragma unroll
    for (int j = 0; j < 8; ++j) {
      const int e = tile * 16 + mb + j;
      __hip_atomic_fetch_add(&agg[(size_t)dstv[e] * DD + n], acc2[j] + bias,
                             __ATOMIC_RELAXED, __HIP_MEMORY_SCOPE_AGENT);
    }
  }
}

// ---------------- 2) node MLP + 2x LayerNorm -> h_local ----------------
__global__ __launch_bounds__(256) void k_node(
    const __bf16* __restrict__ xbf, const float* __restrict__ agg,
    const __bf16* __restrict__ w1, const float* __restrict__ b1,
    const __bf16* __restrict__ w2, const float* __restrict__ b2,
    const float* __restrict__ x,
    const float* __restrict__ gg, const float* __restrict__ gb,
    const float* __restrict__ lg, const float* __restrict__ lb,
    float* __restrict__ hlocal) {
  __shared__ __bf16 A2[16][2 * DD];
  __shared__ __bf16 Hs[16][DD];
  __shared__ float  T[16][DD];
  const int tile = blockIdx.x, tid = threadIdx.x;
  const int lane = tid & 31, wave = tid >> 5;
  {
    const int r  = tid >> 4;
    const int cb = (tid & 15) * 8;
    const int row = tile * 16 + r;
    *(v8bf*)&A2[r][cb] = *(const v8bf*)(xbf + (size_t)row * DD + cb);
    v8bf t;
#pragma unroll
    for (int i = 0; i < 8; ++i) t[i] = f2bf(agg[(size_t)row * DD + cb + i]);
    *(v8bf*)&A2[r][DD + cb] = t;
  }
  __syncthreads();
  v8f acc = {};
#pragma unroll
  for (int kk = 0; kk < 8; ++kk)
    acc = WMMA(load_A(&A2[0][0], 2 * DD, kk * 32, lane), load_Bp(w1, 8, kk, wave, lane), acc);
  {
    const int n = wave * 16 + (lane & 15), mb = (lane >> 4) << 3;
    const float bias = b1[n];
#pragma unroll
    for (int j = 0; j < 8; ++j) {
      float v = acc[j] + bias;
      Hs[mb + j][n] = f2bf(v > 0.f ? v : 0.f);
    }
  }
  __syncthreads();
  v8f acc2 = {};
#pragma unroll
  for (int kk = 0; kk < 4; ++kk)
    acc2 = WMMA(load_A(&Hs[0][0], DD, kk * 32, lane), load_Bp(w2, 4, kk, wave, lane), acc2);
  {
    const int n = wave * 16 + (lane & 15), mb = (lane >> 4) << 3;
    const float bias = b2[n];
#pragma unroll
    for (int j = 0; j < 8; ++j) {
      const int row = tile * 16 + mb + j;
      T[mb + j][n] = acc2[j] + bias + x[(size_t)row * DD + n];   // x + hn
    }
  }
  ln_rows(T, gg, gb, tid);                                        // h_gcn
#pragma unroll
  for (int i = 0; i < 8; ++i) {                                   // x + h_gcn
    const int idx = tid * 8 + i, rr = idx >> 7, cc = idx & 127;
    T[rr][cc] += x[(size_t)(tile * 16 + rr) * DD + cc];
  }
  ln_rows(T, lg, lb, tid);                                        // h_local
#pragma unroll
  for (int i = 0; i < 8; ++i) {
    const int idx = tid * 8 + i, rr = idx >> 7, cc = idx & 127;
    hlocal[(size_t)(tile * 16 + rr) * DD + cc] = T[rr][cc];
  }
}

// ---------------- 3) QKV projection (dense batching == identity reshape) ----------------
// q -> [bh][LL][16] (scaled by 1/sqrt(16)); k -> [bh][LP][16]; v -> transposed [bh][16][LP]
__global__ __launch_bounds__(256) void k_qkv(
    const __bf16* __restrict__ xbf, const __bf16* __restrict__ wqkv,
    const float* __restrict__ bqkv,
    __bf16* __restrict__ qb, __bf16* __restrict__ kb, __bf16* __restrict__ vt) {
  const int tile = blockIdx.x, grp = blockIdx.y;   // grp: 0=q 1=k 2=v
  const int tid = threadIdx.x, lane = tid & 31, wave = tid >> 5;
  const __bf16* A = xbf + (size_t)tile * 16 * DD;
  v8f acc = {};
#pragma unroll
  for (int kk = 0; kk < 4; ++kk)
    acc = WMMA(load_A(A, DD, kk * 32, lane), load_Bp(wqkv, 4, kk, grp * 8 + wave, lane), acc);
  const int dh = lane & 15, mb = (lane >> 4) << 3;
  const int h = wave;                               // col = h*16+dh within this 128-col group
  const float bias = bqkv[grp * DD + wave * 16 + dh];
  const float scale = (grp == 0) ? 0.25f : 1.0f;    // fold 1/sqrt(DH) into q
#pragma unroll
  for (int j = 0; j < 8; ++j) {
    const int row = tile * 16 + mb + j;
    const int b = row / LL, l = row % LL;
    const size_t bh = (size_t)b * HH + h;
    const __bf16 val = f2bf((acc[j] + bias) * scale);
    if (grp == 0)      qb[(bh * LL + l) * 16 + dh] = val;
    else if (grp == 1) kb[(bh * LP + l) * 16 + dh] = val;
    else               vt[(bh * 16 + dh) * LP + l] = val;
  }
}

// ---------------- 4) attention: scores -> softmax -> attn@V ----------------
__global__ __launch_bounds__(256) void k_attn(
    const __bf16* __restrict__ qb, const __bf16* __restrict__ kb,
    const __bf16* __restrict__ vt, __bf16* __restrict__ obf) {
  __shared__ __bf16 Aq[16][32];        // q tile, DH padded 16->32
  __shared__ float  S[16][LP];         // scores / probs (500 padded to 512)
  __shared__ __bf16 Sb[16][LP];
  __shared__ float  Op[8][16][16];     // per-wave partial outputs
  const int qt = blockIdx.x, bh = blockIdx.y;
  const int tid = threadIdx.x, lane = tid & 31, wave = tid >> 5;
  const __bf16* q = qb + (size_t)bh * LL * 16;
  const __bf16* k = kb + (size_t)bh * LP * 16;
  const __bf16* v = vt + (size_t)bh * 16 * LP;
  const int l0 = qt * 16;
  const __bf16 zbf = f2bf(0.f);
#pragma unroll
  for (int i = 0; i < 2; ++i) {        // 256 thr * 2 = 512 = 16*32
    const int idx = tid * 2 + i, rr = idx >> 5, cc = idx & 31;
    const int l = l0 + rr;
    Aq[rr][cc] = (cc < 16 && l < LL) ? q[(size_t)l * 16 + cc] : zbf;
  }
  __syncthreads();
  const v16bf aq = load_A(&Aq[0][0], 32, 0, lane);
#pragma unroll
  for (int j = 0; j < 4; ++j) {        // each wave: 4 column chunks of 16
    const int c0 = (wave + 8 * j) * 16;
    // B[d][n] = k[n][d]: contiguous 32B per lane; lanes 16-31 are the K zero-pad
    v16bf b = {};
    if (lane < 16) b = *(const v16bf*)(k + (size_t)(c0 + lane) * 16);
    v8f acc = {};
    acc = WMMA(aq, b, acc);
    store_C(&S[0][0], LP, c0, acc, lane);
  }
  __syncthreads();
  if (tid < 16) {                      // row softmax (q pre-scaled)
    float mx = -1e30f;
    for (int c = 0; c < LL; ++c) mx = fmaxf(mx, S[tid][c]);
    float sum = 0.f;
    for (int c = 0; c < LP; ++c) {
      const float e = (c < LL) ? __expf(S[tid][c] - mx) : 0.f;
      S[tid][c] = e; sum += e;
    }
    const float inv = 1.f / sum;
    for (int c = 0; c < LP; ++c) S[tid][c] *= inv;
  }
  __syncthreads();
#pragma unroll
  for (int i = 0; i < 32; ++i) {       // probs -> bf16
    const int idx = tid + 256 * i, rr = idx >> 9, cc = idx & 511;
    Sb[rr][cc] = f2bf(S[rr][cc]);
  }
  __syncthreads();
  v8f acc = {};
#pragma unroll
  for (int j = 0; j < 2; ++j) {        // attn @ V, K split across 8 waves
    const int k0 = wave * 64 + j * 32;
    // B[kc][n] = v[kc][n] = vt[n][kc]: contiguous 32B per lane, padding pre-zeroed
    const int kbase = k0 + ((lane >> 4) << 4);
    v16bf b = *(const v16bf*)(v + (size_t)(lane & 15) * LP + kbase);
    acc = WMMA(load_A(&Sb[0][0], LP, k0, lane), b, acc);
  }
  store_C(&Op[wave][0][0], 16, 0, acc, lane);
  __syncthreads();
  {                                    // reduce 8 partials, scatter to [N, D]
    const int rr = tid >> 4, cc = tid & 15;
    float s = 0.f;
#pragma unroll
    for (int w = 0; w < 8; ++w) s += Op[w][rr][cc];
    const int l = l0 + rr;
    if (l < LL) {
      const int b = bh >> 3, h = bh & 7;
      obf[(size_t)(b * LL + l) * DD + h * 16 + cc] = f2bf(s);
    }
  }
}

// ---------------- 5) out-proj + norm_attn + combine ----------------
__global__ __launch_bounds__(256) void k_oproj(
    const __bf16* __restrict__ obf, const __bf16* __restrict__ wo,
    const float* __restrict__ bo, const float* __restrict__ x,
    const float* __restrict__ lg, const float* __restrict__ lb,
    const float* __restrict__ hlocal, float* __restrict__ hf, __bf16* __restrict__ hbf) {
  __shared__ float T[16][DD];
  const int tile = blockIdx.x, tid = threadIdx.x, lane = tid & 31, wave = tid >> 5;
  const __bf16* A = obf + (size_t)tile * 16 * DD;
  v8f acc = {};
#pragma unroll
  for (int kk = 0; kk < 4; ++kk)
    acc = WMMA(load_A(A, DD, kk * 32, lane), load_Bp(wo, 4, kk, wave, lane), acc);
  {
    const int n = wave * 16 + (lane & 15), mb = (lane >> 4) << 3;
    const float bias = bo[n];
#pragma unroll
    for (int j = 0; j < 8; ++j) {
      const int row = tile * 16 + mb + j;
      T[mb + j][n] = acc[j] + bias + x[(size_t)row * DD + n];    // x + h_attn_raw
    }
  }
  ln_rows(T, lg, lb, tid);                                       // h_attn
#pragma unroll
  for (int i = 0; i < 8; ++i) {                                  // h = h_local + h_attn
    const int idx = tid * 8 + i, rr = idx >> 7, cc = idx & 127;
    const size_t g = (size_t)(tile * 16 + rr) * DD + cc;
    const float hv = T[rr][cc] + hlocal[g];
    hf[g] = hv;
    hbf[g] = f2bf(hv);
  }
}

// ---------------- 6) FFN (fused) + final LayerNorm -> out ----------------
__global__ __launch_bounds__(256) void k_ffn(
    const __bf16* __restrict__ hbf, const float* __restrict__ hf,
    const __bf16* __restrict__ w1, const float* __restrict__ b1,
    const __bf16* __restrict__ w2, const float* __restrict__ b2,
    const float* __restrict__ lg, const float* __restrict__ lb,
    float* __restrict__ out) {
  __shared__ __bf16 Hh[16][2 * DD];
  __shared__ float  T[16][DD];
  const int tile = blockIdx.x, tid = threadIdx.x, lane = tid & 31, wave = tid >> 5;
  const __bf16* A = hbf + (size_t)tile * 16 * DD;
#pragma unroll
  for (int g = 0; g < 2; ++g) {        // 256 hidden cols, 2 chunks per wave
    const int nc = g * 8 + wave;
    v8f acc = {};
#pragma unroll
    for (int kk = 0; kk < 4; ++kk)
      acc = WMMA(load_A(A, DD, kk * 32, lane), load_Bp(w1, 4, kk, nc, lane), acc);
    const int n = nc * 16 + (lane & 15), mb = (lane >> 4) << 3;
    const float bias = b1[n];
#pragma unroll
    for (int j = 0; j < 8; ++j) {
      float v = acc[j] + bias;
      Hh[mb + j][n] = f2bf(v > 0.f ? v : 0.f);
    }
  }
  __syncthreads();
  v8f acc = {};
#pragma unroll
  for (int kk = 0; kk < 8; ++kk)
    acc = WMMA(load_A(&Hh[0][0], 2 * DD, kk * 32, lane), load_Bp(w2, 8, kk, wave, lane), acc);
  {
    const int n = wave * 16 + (lane & 15), mb = (lane >> 4) << 3;
    const float bias = b2[n];
#pragma unroll
    for (int j = 0; j < 8; ++j) {
      const int row = tile * 16 + mb + j;
      T[mb + j][n] = acc[j] + bias + hf[(size_t)row * DD + n];   // h + ffn
    }
  }
  ln_rows(T, lg, lb, tid);
#pragma unroll
  for (int i = 0; i < 8; ++i) {
    const int idx = tid * 8 + i, rr = idx >> 7, cc = idx & 127;
    out[(size_t)(tile * 16 + rr) * DD + cc] = T[rr][cc];
  }
}

// ---------------- workspace layout (bytes) ----------------
static constexpr size_t OFF_EW1  = 0;                                   // 384x128 bf16 (packed)
static constexpr size_t OFF_EW2  = OFF_EW1  + (size_t)384 * 128 * 2;
static constexpr size_t OFF_NW1  = OFF_EW2  + (size_t)128 * 128 * 2;
static constexpr size_t OFF_NW2  = OFF_NW1  + (size_t)256 * 128 * 2;
static constexpr size_t OFF_WQKV = OFF_NW2  + (size_t)128 * 128 * 2;
static constexpr size_t OFF_WO   = OFF_WQKV + (size_t)128 * 384 * 2;
static constexpr size_t OFF_FW1  = OFF_WO   + (size_t)128 * 128 * 2;
static constexpr size_t OFF_FW2  = OFF_FW1  + (size_t)128 * 256 * 2;
static constexpr size_t OFF_XBF  = OFF_FW2  + (size_t)256 * 128 * 2;
static constexpr size_t OFF_AGG  = OFF_XBF  + (size_t)NN * DD * 2;
static constexpr size_t OFF_HLOC = OFF_AGG  + (size_t)NN * DD * 4;
static constexpr size_t OFF_QBF  = OFF_HLOC + (size_t)NN * DD * 4;
static constexpr size_t OFF_KBF  = OFF_QBF  + (size_t)BH * LL * 16 * 2;
static constexpr size_t OFF_VT   = OFF_KBF  + (size_t)BH * LP * 16 * 2;
static constexpr size_t OFF_OBF  = OFF_VT   + (size_t)BH * 16 * LP * 2;
static constexpr size_t OFF_HF   = OFF_OBF  + (size_t)NN * DD * 2;
static constexpr size_t OFF_HBF  = OFF_HF   + (size_t)NN * DD * 4;

static inline __bf16* wsbf(void* ws, size_t off) { return (__bf16*)((char*)ws + off); }
static inline float*  wsf (void* ws, size_t off) { return (float*)((char*)ws + off); }

extern "C" void kernel_launch(void* const* d_in, const int* in_sizes, int n_in,
                              void* d_out, int out_size, void* d_ws, size_t ws_size,
                              hipStream_t stream) {
  // setup_inputs() dict order: x, edge_attr, params{insertion order}, edge_index, batch
  const float* x     = (const float*)d_in[0];
  const float* eattr = (const float*)d_in[1];
  const float* e_w1  = (const float*)d_in[2];
  const float* e_b1  = (const float*)d_in[3];
  const float* e_w2  = (const float*)d_in[4];
  const float* e_b2  = (const float*)d_in[5];
  const float* n_w1  = (const float*)d_in[6];
  const float* n_b1  = (const float*)d_in[7];
  const float* n_w2  = (const float*)d_in[8];
  const float* n_b2  = (const float*)d_in[9];
  const float* gcn_g = (const float*)d_in[10];
  const float* gcn_b = (const float*)d_in[11];
  const float* lnl_g = (const float*)d_in[12];
  const float* lnl_b = (const float*)d_in[13];
  const float* lna_g = (const float*)d_in[14];
  const float* lna_b = (const float*)d_in[15];
  const float* lnf_g = (const float*)d_in[16];
  const float* lnf_b = (const float*)d_in[17];
  const float* wqkv  = (const float*)d_in[18];
  const float* bqkv  = (const float*)d_in[19];
  const float* wo    = (const float*)d_in[20];
  const float* bo    = (const float*)d_in[21];
  const float* f_w1  = (const float*)d_in[22];
  const float* f_b1  = (const float*)d_in[23];
  const float* f_w2  = (const float*)d_in[24];
  const float* f_b2  = (const float*)d_in[25];
  const int*   eidx  = (const int*)d_in[26];
  float* out = (float*)d_out;

  // ---- prep: pack weights to B-fragment order, x -> bf16, zero agg/k/v pads ----
  struct { const float* s; size_t off; int K, Nw; } pk[] = {
    { e_w1, OFF_EW1,  384, 128 }, { e_w2, OFF_EW2,  128, 128 },
    { n_w1, OFF_NW1,  256, 128 }, { n_w2, OFF_NW2,  128, 128 },
    { wqkv, OFF_WQKV, 128, 384 }, { wo,   OFF_WO,   128, 128 },
    { f_w1, OFF_FW1,  128, 256 }, { f_w2, OFF_FW2,  256, 128 },
  };
  for (auto& c : pk)
    k_pack_b<<<(c.K * c.Nw + 255) / 256, 256, 0, stream>>>(c.s, wsbf(d_ws, c.off), c.K, c.Nw);
  k_cvt_bf16<<<(NN * DD + 255) / 256, 256, 0, stream>>>(x, wsbf(d_ws, OFF_XBF), NN * DD);
  k_zero_f32<<<(NN * DD + 255) / 256, 256, 0, stream>>>(wsf(d_ws, OFF_AGG), NN * DD);
  k_zero_bf16<<<(BH * LP * 16 + 255) / 256, 256, 0, stream>>>(wsbf(d_ws, OFF_KBF), BH * LP * 16);
  k_zero_bf16<<<(BH * 16 * LP + 255) / 256, 256, 0, stream>>>(wsbf(d_ws, OFF_VT),  BH * 16 * LP);

  // ---- 1) edge MLP + scatter ----
  k_edge<<<NE / 16, 256, 0, stream>>>(wsbf(d_ws, OFF_XBF), eattr, eidx,
                                      wsbf(d_ws, OFF_EW1), e_b1,
                                      wsbf(d_ws, OFF_EW2), e_b2,
                                      wsf(d_ws, OFF_AGG));
  // ---- 2) node MLP + norms -> h_local ----
  k_node<<<NN / 16, 256, 0, stream>>>(wsbf(d_ws, OFF_XBF), wsf(d_ws, OFF_AGG),
                                      wsbf(d_ws, OFF_NW1), n_b1,
                                      wsbf(d_ws, OFF_NW2), n_b2,
                                      x, gcn_g, gcn_b, lnl_g, lnl_b,
                                      wsf(d_ws, OFF_HLOC));
  // ---- 3) QKV ----
  k_qkv<<<dim3(NN / 16, 3), 256, 0, stream>>>(wsbf(d_ws, OFF_XBF), wsbf(d_ws, OFF_WQKV), bqkv,
                                              wsbf(d_ws, OFF_QBF), wsbf(d_ws, OFF_KBF),
                                              wsbf(d_ws, OFF_VT));
  // ---- 4) attention ----
  k_attn<<<dim3((LL + 15) / 16, BH), 256, 0, stream>>>(wsbf(d_ws, OFF_QBF), wsbf(d_ws, OFF_KBF),
                                                       wsbf(d_ws, OFF_VT), wsbf(d_ws, OFF_OBF));
  // ---- 5) out-proj + norm + combine ----
  k_oproj<<<NN / 16, 256, 0, stream>>>(wsbf(d_ws, OFF_OBF), wsbf(d_ws, OFF_WO), bo,
                                       x, lna_g, lna_b, wsf(d_ws, OFF_HLOC),
                                       wsf(d_ws, OFF_HF), wsbf(d_ws, OFF_HBF));
  // ---- 6) FFN + final LN ----
  k_ffn<<<NN / 16, 256, 0, stream>>>(wsbf(d_ws, OFF_HBF), wsf(d_ws, OFF_HF),
                                     wsbf(d_ws, OFF_FW1), f_b1,
                                     wsbf(d_ws, OFF_FW2), f_b2,
                                     lnf_g, lnf_b, out);
  (void)in_sizes; (void)n_in; (void)out_size; (void)ws_size;
}
